// DeepSeekV3LatentAttention_26001732010570
// MI455X (gfx1250) — matrix operand order repaired and checked by reading, not compiled, not run on hardware
//
#include <hip/hip_runtime.h>

// ---------------- problem constants ----------------
#define BB   2
#define TT   2048
#define HH   16
#define HD   128
#define RD   64
#define LAT  512
#define DIN  2048
#define DOUT 2048
#define DQ   3072          // D_OUT + H*RD
#define QROPE 1024
#define SCALE 0.07216878365f   // 1/sqrt(HD+RD) = 1/sqrt(192)

typedef unsigned short u16;
typedef unsigned int   u32;
typedef __attribute__((ext_vector_type(16))) __bf16 v16bf;
typedef __attribute__((ext_vector_type(8)))  float  v8f;

union Frag { v16bf v; uint4 q[2]; };
union Pack8 { uint4 q; u16 h[8]; };

__device__ __forceinline__ u16 f2bf(float f) {
    u32 u = __float_as_uint(f);
    u32 r = (u + 0x7fffu + ((u >> 16) & 1u)) >> 16;
    return (u16)r;
}

__device__ __forceinline__ v8f wmma_bf16(const Frag& a, const Frag& b, v8f c) {
    return __builtin_amdgcn_wmma_f32_16x16x32_bf16(
        false, a.v, false, b.v, (short)0, c, false, false);
}

// ---- CDNA5 async global->LDS (16B per lane), tracked by ASYNCcnt ----
// LDS dest address = low 32 bits of the generic shared pointer (ISA 10.2:
// flat->LDS mapping uses addr[31:0] as the LDS offset).
__device__ __forceinline__ void async_ld16(void* ldsPtr, const void* gptr) {
    unsigned lo = (unsigned)(unsigned long long)(uintptr_t)ldsPtr;
    unsigned long long ga = (unsigned long long)(uintptr_t)gptr;
    asm volatile("global_load_async_to_lds_b128 %0, %1, off"
                 :: "v"(lo), "v"(ga) : "memory");
}
__device__ __forceinline__ void wait_async0() {
    asm volatile("s_wait_asynccnt 0" ::: "memory");
}

// A fragment: 16x32 bf16, rowPtr = &A[row*lda + k0]; kbA = (lane>=16)?8:0
__device__ __forceinline__ void loadA(Frag& f, const u16* rowPtr, int kbA) {
    f.q[0] = *(const uint4*)(rowPtr + kbA);
    f.q[1] = *(const uint4*)(rowPtr + kbA + 16);
}
// B fragment from N-major W[n][k]: rowPtr = &W[n*ldb + k0]; kbB = (lane>=16)?16:0
__device__ __forceinline__ void loadB(Frag& f, const u16* rowPtr, int kbB) {
    const uint4* p = (const uint4*)(rowPtr + (kbB >> 1));
    f.q[0] = p[0];
    f.q[1] = p[2];
}

// ---------------- elementwise converts ----------------
__global__ __launch_bounds__(256) void cvt2d_kernel(
    const float* __restrict__ src, long sld,
    u16* __restrict__ dst, long dld, long rows, long cols)
{
    long idx = (long)blockIdx.x * 256 + threadIdx.x;
    long total = rows * cols;
    if (idx >= total) return;
    long r = idx / cols, c = idx % cols;
    dst[r * dld + c] = f2bf(src[r * sld + c]);
}

// W_UK [2048,512] -> W_UKt bf16 [H][LAT][HD]
__global__ __launch_bounds__(256) void wukt_kernel(
    const float* __restrict__ wuk, u16* __restrict__ out)
{
    long idx = (long)blockIdx.x * 256 + threadIdx.x;
    if (idx >= (long)DOUT * LAT) return;
    int d = (int)(idx % HD);
    int l = (int)((idx / HD) % LAT);
    int h = (int)(idx / ((long)HD * LAT));
    out[idx] = f2bf(wuk[(long)(h * HD + d) * LAT + l]);
}

// ---------------- RMSNorm -> bf16 ----------------
__global__ __launch_bounds__(256) void rmsnorm_kernel(
    const float* __restrict__ x, const float* __restrict__ w, u16* __restrict__ out)
{
    __shared__ float red[256];
    int tid = threadIdx.x;
    long r = blockIdx.x;
    const float* xr = x + r * LAT;
    float s = 0.f;
    for (int c = tid; c < LAT; c += 256) { float v = xr[c]; s += v * v; }
    red[tid] = s; __syncthreads();
    for (int o = 128; o > 0; o >>= 1) {
        if (tid < o) red[tid] += red[tid + o];
        __syncthreads();
    }
    float sc = rsqrtf(red[0] / (float)LAT + 1e-6f);
    for (int c = tid; c < LAT; c += 256)
        out[r * LAT + c] = f2bf(xr[c] * sc * w[c]);
}

// ---------------- RoPE (q and k), output bf16 [B,H,T,RD] ----------------
__global__ __launch_bounds__(256) void rope_kernel(
    const float* __restrict__ q_total, const float* __restrict__ krr,
    const int* __restrict__ offp,
    u16* __restrict__ qr_b, u16* __restrict__ kr_b)
{
    long idx = (long)blockIdx.x * 256 + threadIdx.x;   // B*H*T*RD
    if (idx >= (long)BB * HH * TT * RD) return;
    int r = (int)(idx & (RD - 1));
    long rem = idx >> 6;
    int t = (int)(rem % TT);
    long bh = rem / TT;
    int h = (int)(bh % HH);
    int b = (int)(bh / HH);
    int rr = r & 31;
    float inv = __powf(10000.f, -((float)(2 * rr) / (float)RD));
    float pos = (float)(t + *offp);
    float sn, cs;
    __sincosf(pos * inv, &sn, &cs);
    long row = (long)b * TT + t;
    const float* qrow = q_total + row * DQ + DOUT + (long)h * RD;
    const float* krow = krr + row * QROPE + (long)h * RD;
    float qv = qrow[r], kv = krow[r];
    float qrot = (r < 32) ? -qrow[r + 32] : qrow[r - 32];
    float krot = (r < 32) ? -krow[r + 32] : krow[r - 32];
    qr_b[idx] = f2bf(qv * cs + qrot * sn);
    kr_b[idx] = f2bf(kv * cs + krot * sn);
}

// ---------------- generic WMMA GEMM: C[M,N](f32|bf16) = A[M,K]bf16 * B[N,K]bf16^T (+bias) ----------------
// Double-buffered async global->LDS staging: stage k0+32 loads overlap stage k0 compute.
__global__ __launch_bounds__(256) void gemm_bf16_kernel(
    const u16* __restrict__ A, long lda, long aStrideZ,
    const u16* __restrict__ B, long ldb, long bStrideZ,
    void* __restrict__ Cv, long ldc, long cStrideZ,
    int M, int N, int K,
    const float* __restrict__ bias, int outBf16)
{
    __shared__ __align__(16) u16 sA[2][128 * 32];
    __shared__ __align__(16) u16 sB[2][128 * 32];
    int z = blockIdx.z;
    A += (long)z * aStrideZ;
    B += (long)z * bStrideZ;
    long cBase = (long)z * cStrideZ;
    int m0 = blockIdx.y * 128, n0 = blockIdx.x * 128;
    int tid = threadIdx.x, w = tid >> 5, lane = tid & 31;
    int l = lane & 15;
    int kbA = (lane >> 4) * 8;
    int kbB = (lane >> 4) * 16;
    int mw = (w >> 1) * 32;      // wave row base in tile
    int nw = (w & 1) * 64;       // wave col base in tile

    // this thread's two 8-element chunks of the 128x32 tile
    int c0 = tid * 2, c1 = tid * 2 + 1;
    int r0 = c0 >> 2, col0 = (c0 & 3) * 8;
    int r1 = c1 >> 2, col1 = (c1 & 3) * 8;

    v8f acc[2][4] = {};
    int nst = K / 32;

    // prologue: stage 0
    async_ld16(&sA[0][r0 * 32 + col0], &A[(long)(m0 + r0) * lda + col0]);
    async_ld16(&sA[0][r1 * 32 + col1], &A[(long)(m0 + r1) * lda + col1]);
    async_ld16(&sB[0][r0 * 32 + col0], &B[(long)(n0 + r0) * ldb + col0]);
    async_ld16(&sB[0][r1 * 32 + col1], &B[(long)(n0 + r1) * ldb + col1]);
    wait_async0();
    __syncthreads();

    for (int st = 0; st < nst; ++st) {
        int cur = st & 1;
        if (st + 1 < nst) {
            int k0 = (st + 1) * 32;
            int nxt = cur ^ 1;
            async_ld16(&sA[nxt][r0 * 32 + col0], &A[(long)(m0 + r0) * lda + k0 + col0]);
            async_ld16(&sA[nxt][r1 * 32 + col1], &A[(long)(m0 + r1) * lda + k0 + col1]);
            async_ld16(&sB[nxt][r0 * 32 + col0], &B[(long)(n0 + r0) * ldb + k0 + col0]);
            async_ld16(&sB[nxt][r1 * 32 + col1], &B[(long)(n0 + r1) * ldb + k0 + col1]);
        }
        Frag af[2], bfm[4];
        for (int i = 0; i < 2; ++i) loadA(af[i], &sA[cur][(mw + i * 16 + l) * 32], kbA);
        for (int j = 0; j < 4; ++j) loadB(bfm[j], &sB[cur][(nw + j * 16 + l) * 32], kbB);
        for (int i = 0; i < 2; ++i)
            for (int j = 0; j < 4; ++j)
                acc[i][j] = wmma_bf16(af[i], bfm[j], acc[i][j]);
        wait_async0();
        __syncthreads();
    }
    int rsel = (lane >> 4) * 8;
    for (int i = 0; i < 2; ++i)
        for (int j = 0; j < 4; ++j) {
            int col = n0 + nw + j * 16 + l;
            float bv = bias ? bias[col] : 0.f;
            for (int e = 0; e < 8; ++e) {
                int row = m0 + mw + i * 16 + rsel + e;
                float v = acc[i][j][e] + bv;
                if (outBf16) ((u16*)Cv)[cBase + (long)row * ldc + col] = f2bf(v);
                else         ((float*)Cv)[cBase + (long)row * ldc + col] = v;
            }
        }
}

// ---------------- attention pass 1: row max / sum-exp ----------------
// grid: (T/64, H, B), 256 threads. q-block 64 rows; s-tiles of 32.
__global__ __launch_bounds__(256) void attn_stats_kernel(
    const u16* __restrict__ qabs, const u16* __restrict__ qr,
    const u16* __restrict__ ckv,  const u16* __restrict__ kr,
    float* __restrict__ mbuf, float* __restrict__ lbuf)
{
    __shared__ __align__(16) u16 sK[32 * LAT];      // 32KB
    __shared__ __align__(16) u16 sKr[32 * RD];      // 4KB
    __shared__ float sS[64 * 32];                   // 8KB
    int qb = blockIdx.x, h = blockIdx.y, b = blockIdx.z;
    int q0 = qb * 64;
    int tid = threadIdx.x, w = tid >> 5, lane = tid & 31;
    int l = lane & 15, kbA = (lane >> 4) * 8, kbB = (lane >> 4) * 16;
    int rt = (w & 3) * 16, ct = (w >> 2) * 16;
    const u16* qa  = qabs + ((long)(b * HH + h) * TT + q0) * LAT;
    const u16* qrp = qr   + ((long)(b * HH + h) * TT + q0) * RD;
    const u16* kvb = ckv  + (long)b * TT * LAT;
    const u16* krp = kr   + ((long)(b * HH + h) * TT) * RD;

    float m = -__builtin_inff(), lsum = 0.f;
    int nS = (q0 + 64 + 31) / 32;
    for (int st = 0; st < nS; ++st) {
        int s0 = st * 32;
        for (int i = 0; i < 8; ++i) {                 // 32x512 bf16, async to LDS
            int c = tid + i * 256;
            int rr = c >> 6, col = (c & 63) * 8;
            async_ld16(&sK[rr * LAT + col], &kvb[(long)(s0 + rr) * LAT + col]);
        }
        {                                             // 32x64 bf16
            int rr = tid >> 3, col = (tid & 7) * 8;
            async_ld16(&sKr[rr * RD + col], &krp[(long)(s0 + rr) * RD + col]);
        }
        wait_async0();
        __syncthreads();
        v8f acc = {};
        const u16* arow = qa + (long)(rt + l) * LAT;
        for (int k0 = 0; k0 < LAT; k0 += 32) {
            Frag a, bm;
            loadA(a, arow + k0, kbA);
            loadB(bm, &sK[(ct + l) * LAT + k0], kbB);
            acc = wmma_bf16(a, bm, acc);
        }
        const u16* arow2 = qrp + (long)(rt + l) * RD;
        for (int k0 = 0; k0 < RD; k0 += 32) {
            Frag a, bm;
            loadA(a, arow2 + k0, kbA);
            loadB(bm, &sKr[(ct + l) * RD + k0], kbB);
            acc = wmma_bf16(a, bm, acc);
        }
        int rsel = (lane >> 4) * 8;
        for (int e = 0; e < 8; ++e)
            sS[(rt + rsel + e) * 32 + ct + l] = acc[e];
        __syncthreads();
        if (tid < 64) {
            int r = tid, qrow = q0 + r;
            float mnew = m;
            for (int c = 0; c < 32; ++c) {
                int s = s0 + c;
                if (s <= qrow) {
                    float v = sS[r * 32 + c] * SCALE;
                    mnew = fmaxf(mnew, v);
                }
            }
            if (mnew > -__builtin_inff()) {
                float corr = (m > -__builtin_inff()) ? __expf(m - mnew) : 0.f;
                float add = 0.f;
                for (int c = 0; c < 32; ++c) {
                    int s = s0 + c;
                    if (s <= qrow) add += __expf(sS[r * 32 + c] * SCALE - mnew);
                }
                lsum = lsum * corr + add;
                m = mnew;
            }
        }
        __syncthreads();
    }
    if (tid < 64) {
        long idx = (long)(b * HH + h) * TT + q0 + tid;
        mbuf[idx] = m;
        lbuf[idx] = lsum;
    }
}

// ---------------- attention pass 2: P@V + fused W_UV, ctx -> bf16 [B,T,DOUT] ----------------
// grid: (T/32, H, B), 256 threads. q-block 32 rows; wave w owns LAT cols [w*64, w*64+64)
__global__ __launch_bounds__(256) void attn_pv_kernel(
    const u16* __restrict__ qabs, const u16* __restrict__ qr,
    const u16* __restrict__ ckv,  const u16* __restrict__ kr,
    const float* __restrict__ mbuf, const float* __restrict__ lbuf,
    const u16* __restrict__ wuv, u16* __restrict__ ctx)
{
    __shared__ __align__(16) u16 sKT[LAT * 32];     // V^T tile [l][s]; reused as O[32][512]
    __shared__ __align__(16) u16 sP[32 * 32];
    __shared__ float sS[32 * 32];
    __shared__ float sM[32], sInvL[32];
    int qb = blockIdx.x, h = blockIdx.y, b = blockIdx.z;
    int q0 = qb * 32;
    int tid = threadIdx.x, w = tid >> 5, lane = tid & 31;
    int l = lane & 15, kbA = (lane >> 4) * 8, kbB = (lane >> 4) * 16;
    const u16* qa  = qabs + ((long)(b * HH + h) * TT + q0) * LAT;
    const u16* qrp = qr   + ((long)(b * HH + h) * TT + q0) * RD;
    const u16* kvb = ckv  + (long)b * TT * LAT;
    const u16* krp = kr   + ((long)(b * HH + h) * TT) * RD;
    if (tid < 32) {
        long idx = (long)(b * HH + h) * TT + q0 + tid;
        sM[tid] = mbuf[idx];
        float lv = lbuf[idx];
        sInvL[tid] = (lv > 0.f) ? 1.f / lv : 0.f;
    }
    __syncthreads();

    v8f o[2][4] = {};
    int nS = qb + 1;
    for (int st = 0; st < nS; ++st) {
        int s0 = st * 32;
        for (int i = 0; i < 8; ++i) {               // load+transpose KV tile into sKT[l][s]
            int c = tid + i * 256;
            int s = c >> 6, colb = (c & 63) * 8;
            Pack8 t;
            t.q = *(const uint4*)&kvb[(long)(s0 + s) * LAT + colb];
            for (int j = 0; j < 8; ++j)
                sKT[(colb + j) * 32 + s] = t.h[j];
        }
        __syncthreads();
        if (w < 4) {                                // scores S[32x32], waves 0..3
            int rt = (w & 1) * 16, ct = (w >> 1) * 16;
            v8f acc = {};
            const u16* arow = qa + (long)(rt + l) * LAT;
            const u16* brow = kvb + (long)(s0 + ct + l) * LAT;
            for (int k0 = 0; k0 < LAT; k0 += 32) {
                Frag a, bm;
                loadA(a, arow + k0, kbA);
                loadB(bm, brow + k0, kbB);
                acc = wmma_bf16(a, bm, acc);
            }
            const u16* arow2 = qrp + (long)(rt + l) * RD;
            const u16* brow2 = krp + (long)(s0 + ct + l) * RD;
            for (int k0 = 0; k0 < RD; k0 += 32) {
                Frag a, bm;
                loadA(a, arow2 + k0, kbA);
                loadB(bm, brow2 + k0, kbB);
                acc = wmma_bf16(a, bm, acc);
            }
            int rsel = (lane >> 4) * 8;
            for (int e = 0; e < 8; ++e)
                sS[(rt + rsel + e) * 32 + ct + l] = acc[e];
        }
        __syncthreads();
        for (int i = 0; i < 4; ++i) {               // P = exp(scale*S - m)/l, masked
            int idx = i * 256 + tid;
            int r = idx >> 5, c = idx & 31;
            int s = s0 + c, qrow = q0 + r;
            float p = 0.f;
            if (s <= qrow) p = __expf(sS[idx] * SCALE - sM[r]) * sInvL[r];
            sP[idx] = f2bf(p);
        }
        __syncthreads();
        for (int i = 0; i < 2; ++i) {               // O += P @ V  (K=32)
            Frag a;
            loadA(a, &sP[(i * 16 + l) * 32], kbA);
            for (int j = 0; j < 4; ++j) {
                Frag bm;
                loadB(bm, &sKT[(w * 64 + j * 16 + l) * 32], kbB);
                o[i][j] = wmma_bf16(a, bm, o[i][j]);
            }
        }
        __syncthreads();
    }
    // dump O (bf16) into sKT reused as [32][512]
    {
        int rsel = (lane >> 4) * 8;
        for (int i = 0; i < 2; ++i)
            for (int j = 0; j < 4; ++j)
                for (int e = 0; e < 8; ++e) {
                    int row = i * 16 + rsel + e;
                    int col = w * 64 + j * 16 + l;
                    sKT[row * LAT + col] = f2bf(o[i][j][e]);
                }
    }
    __syncthreads();
    // ctx[32 x 128] = O(32x512) * W_UV_h(128x512)^T ; 16 tiles, 2 per wave
    const u16* wuvh = wuv + (long)h * HD * LAT;
    for (int tt = 0; tt < 2; ++tt) {
        int t = w * 2 + tt;
        int rt = (t >> 3) * 16, ct = (t & 7) * 16;
        v8f acc = {};
        const u16* arow = &sKT[(rt + l) * LAT];
        const u16* brow = wuvh + (long)(ct + l) * LAT;
        for (int k0 = 0; k0 < LAT; k0 += 32) {
            Frag a, bm;
            loadA(a, arow + k0, kbA);
            loadB(bm, brow + k0, kbB);
            acc = wmma_bf16(a, bm, acc);
        }
        int rsel = (lane >> 4) * 8;
        for (int e = 0; e < 8; ++e) {
            int row = q0 + rt + rsel + e;
            int col = h * HD + ct + l;
            ctx[((long)b * TT + row) * DOUT + col] = f2bf(acc[e]);
        }
    }
}

// ---------------- host launch ----------------
extern "C" void kernel_launch(void* const* d_in, const int* in_sizes, int n_in,
                              void* d_out, int out_size, void* d_ws, size_t ws_size,
                              hipStream_t stream) {
    (void)in_sizes; (void)n_in; (void)out_size; (void)ws_size;
    const float* x      = (const float*)d_in[0];
    const float* W_DKV  = (const float*)d_in[1];
    const float* kvw    = (const float*)d_in[2];
    const float* W_KR   = (const float*)d_in[3];
    const float* W_Q    = (const float*)d_in[4];
    const float* W_UK   = (const float*)d_in[5];
    const float* W_UV   = (const float*)d_in[6];
    const float* out_w  = (const float*)d_in[7];
    const float* out_b  = (const float*)d_in[8];
    const int*   offp   = (const int*)d_in[9];
    float* out = (float*)d_out;

    const long MR = (long)BB * TT;          // 4096 rows
    size_t off = 0;
    auto carve = [&](size_t bytes) -> char* {
        char* p = (char*)d_ws + off;
        off += (bytes + 255) & ~(size_t)255;
        return p;
    };
    u16*   xb       = (u16*)  carve(MR * DIN * 2);
    u16*   wdkv_b   = (u16*)  carve((size_t)LAT * DIN * 2);
    u16*   wkr_b    = (u16*)  carve((size_t)QROPE * DIN * 2);
    u16*   wq_b     = (u16*)  carve((size_t)DQ * DIN * 2);
    u16*   wuv_b    = (u16*)  carve((size_t)DOUT * LAT * 2);
    u16*   outw_b   = (u16*)  carve((size_t)DOUT * DOUT * 2);
    u16*   wukt_b   = (u16*)  carve((size_t)DOUT * LAT * 2);
    float* ckv_raw  = (float*)carve(MR * LAT * 4);
    float* kr_raw   = (float*)carve(MR * QROPE * 4);
    float* q_total  = (float*)carve(MR * DQ * 4);
    u16*   qc_b     = (u16*)  carve(MR * DOUT * 2);
    u16*   ckv_b    = (u16*)  carve(MR * LAT * 2);
    u16*   qabs_b   = (u16*)  carve((size_t)BB * HH * TT * LAT * 2);
    u16*   qr_b     = (u16*)  carve((size_t)BB * HH * TT * RD * 2);
    u16*   kr_b     = (u16*)  carve((size_t)BB * HH * TT * RD * 2);
    float* mbuf     = (float*)carve((size_t)BB * HH * TT * 4);
    float* lbuf     = (float*)carve((size_t)BB * HH * TT * 4);
    u16*   ctx_b    = (u16*)  carve(MR * DOUT * 2);

    auto blocks = [](long n) { return (unsigned)((n + 255) / 256); };

    // bf16 conversions
    cvt2d_kernel<<<blocks(MR * DIN), 256, 0, stream>>>(x, DIN, xb, DIN, MR, DIN);
    cvt2d_kernel<<<blocks((long)LAT * DIN), 256, 0, stream>>>(W_DKV, DIN, wdkv_b, DIN, LAT, DIN);
    cvt2d_kernel<<<blocks((long)QROPE * DIN), 256, 0, stream>>>(W_KR, DIN, wkr_b, DIN, QROPE, DIN);
    cvt2d_kernel<<<blocks((long)DQ * DIN), 256, 0, stream>>>(W_Q, DIN, wq_b, DIN, DQ, DIN);
    cvt2d_kernel<<<blocks((long)DOUT * LAT), 256, 0, stream>>>(W_UV, LAT, wuv_b, LAT, DOUT, LAT);
    cvt2d_kernel<<<blocks((long)DOUT * DOUT), 256, 0, stream>>>(out_w, DOUT, outw_b, DOUT, DOUT, DOUT);
    wukt_kernel<<<blocks((long)DOUT * LAT), 256, 0, stream>>>(W_UK, wukt_b);

    // projections: C = A * B^T
    gemm_bf16_kernel<<<dim3(LAT / 128, MR / 128, 1), 256, 0, stream>>>(
        xb, DIN, 0, wdkv_b, DIN, 0, ckv_raw, LAT, 0, (int)MR, LAT, DIN, nullptr, 0);
    gemm_bf16_kernel<<<dim3(QROPE / 128, MR / 128, 1), 256, 0, stream>>>(
        xb, DIN, 0, wkr_b, DIN, 0, kr_raw, QROPE, 0, (int)MR, QROPE, DIN, nullptr, 0);
    gemm_bf16_kernel<<<dim3(DQ / 128, MR / 128, 1), 256, 0, stream>>>(
        xb, DIN, 0, wq_b, DIN, 0, q_total, DQ, 0, (int)MR, DQ, DIN, nullptr, 0);

    // rmsnorm -> bf16 KV latent
    rmsnorm_kernel<<<(unsigned)MR, 256, 0, stream>>>(ckv_raw, kvw, ckv_b);
    // q content -> bf16 (strided view of q_total)
    cvt2d_kernel<<<blocks(MR * DOUT), 256, 0, stream>>>(q_total, DQ, qc_b, DOUT, MR, DOUT);
    // rope on q/k
    rope_kernel<<<blocks((long)BB * HH * TT * RD), 256, 0, stream>>>(q_total, kr_raw, offp, qr_b, kr_b);

    // q_abs = q_content @ W_UK^T per head (grid.z = head, one launch per batch)
    for (int b = 0; b < BB; ++b) {
        gemm_bf16_kernel<<<dim3(LAT / 128, TT / 128, HH), 256, 0, stream>>>(
            qc_b + (long)b * TT * DOUT, DOUT, HD,
            wukt_b, HD, (long)LAT * HD,
            qabs_b + (long)b * HH * TT * LAT, LAT, (long)TT * LAT,
            TT, LAT, HD, nullptr, 1);
    }

    // attention
    attn_stats_kernel<<<dim3(TT / 64, HH, BB), 256, 0, stream>>>(
        qabs_b, qr_b, ckv_b, kr_b, mbuf, lbuf);
    attn_pv_kernel<<<dim3(TT / 32, HH, BB), 256, 0, stream>>>(
        qabs_b, qr_b, ckv_b, kr_b, mbuf, lbuf, wuv_b, ctx_b);

    // final projection with bias
    gemm_bf16_kernel<<<dim3(DOUT / 128, MR / 128, 1), 256, 0, stream>>>(
        ctx_b, DOUT, 0, outw_b, DOUT, 0, out, DOUT, 0, (int)MR, DOUT, DIN, out_b, 0);
}
